// RodimusAttention_13752485282238
// MI455X (gfx1250) — compile-verified, matrix operand order checked by reading, not run on hardware
//
#include <hip/hip_runtime.h>

// ---------------------------------------------------------------------------
// Types for CDNA5 WMMA (gfx1250, wave32)
// ---------------------------------------------------------------------------
typedef __attribute__((ext_vector_type(16))) __bf16 v16bf;
typedef __attribute__((ext_vector_type(8)))  __bf16 v8bf;
typedef __attribute__((ext_vector_type(8)))  float  v8f;

union BF16Frag { v16bf v; v8bf h[2]; };

__device__ __forceinline__ __bf16 to_bf16(float f) { return (__bf16)f; }

__device__ __forceinline__ v8f wmma_bf16(v16bf a, v16bf b, v8f c) {
  // D = A(16x32) * B(32x16) + C, fp32 accumulate
  return __builtin_amdgcn_wmma_f32_16x16x32_bf16(
      /*neg_a=*/false, a, /*neg_b=*/false, b,
      /*c_mod=*/(short)0, c, /*reuse_a=*/false, /*reuse_b=*/false);
}

// Fragment of a row-major bf16 matrix with leading dim 64 (LDS tiles).
__device__ __forceinline__ v16bf lds_frag64(const __bf16* base, int row0, int k0, int lane) {
  BF16Frag f;
  const __bf16* p = base + (row0 + (lane & 15)) * 64 + k0 + ((lane >> 4) << 3);
  f.h[0] = *(const v8bf*)p;
  f.h[1] = *(const v8bf*)(p + 16);
  return f.v;
}

__device__ __forceinline__ v16bf gmem_frag(const __bf16* p) {
  BF16Frag f;
  f.h[0] = *(const v8bf*)p;
  f.h[1] = *(const v8bf*)(p + 16);
  return f.v;
}

__device__ __forceinline__ float siluf(float x) { return x / (1.f + __expf(-x)); }
__device__ __forceinline__ float sigmf(float x) { return 1.f / (1.f + __expf(-x)); }

// ---------------------------------------------------------------------------
// f32 -> bf16 conversion (grid-stride)
// ---------------------------------------------------------------------------
__global__ void f32_to_bf16(const float* __restrict__ src, __bf16* __restrict__ dst, long n) {
  long i = blockIdx.x * (long)blockDim.x + threadIdx.x;
  long stride = (long)gridDim.x * blockDim.x;
  for (; i < n; i += stride) dst[i] = to_bf16(src[i]);
}

// ---------------------------------------------------------------------------
// GEMM (1x1 tile/wave): C[MxN] f32 = A[MxKd] bf16 * Bw[NxKd]^T. For small N.
// ---------------------------------------------------------------------------
__global__ void gemm_bf16_wmma(const __bf16* __restrict__ A, const __bf16* __restrict__ Bw,
                               float* __restrict__ C, int M, int N, int Kd) {
  int gw   = (int)((blockIdx.x * (long)blockDim.x + threadIdx.x) >> 5);
  int tn   = N >> 4;
  int tm   = gw / tn;
  int tnn  = gw - tm * tn;
  if (tm >= (M >> 4)) return;  // wave-uniform
  int lane = threadIdx.x & 31;
  int m0 = tm << 4, n0 = tnn << 4;

  const __bf16* pa = A  + (long)(m0 + (lane & 15)) * Kd + ((lane >> 4) << 3);
  const __bf16* pb = Bw + (long)(n0 + (lane & 15)) * Kd + ((lane >> 4) << 3);

  v8f acc = {};
  for (int k0 = 0; k0 < Kd; k0 += 32) {
    acc = wmma_bf16(gmem_frag(pa + k0), gmem_frag(pb + k0), acc);
  }
  int col = n0 + (lane & 15);
  int rb  = m0 + ((lane >> 4) << 3);
  for (int r = 0; r < 8; r++) C[(long)(rb + r) * N + col] = acc[r];
}

// ---------------------------------------------------------------------------
// GEMM (2x2 tiles/wave = 32x32 C block): doubles FLOP/byte vs 1x1 and gives
// four independent WMMA accumulation chains per wave. Requires M%32==N%32==0.
// ---------------------------------------------------------------------------
__global__ void gemm_bf16_wmma2(const __bf16* __restrict__ A, const __bf16* __restrict__ Bw,
                                float* __restrict__ C, int M, int N, int Kd) {
  int gw   = (int)((blockIdx.x * (long)blockDim.x + threadIdx.x) >> 5);
  int tn   = N >> 5;
  int tm   = gw / tn;
  int tnn  = gw - tm * tn;
  if (tm >= (M >> 5)) return;  // wave-uniform
  int lane = threadIdx.x & 31;
  int m0 = tm << 5, n0 = tnn << 5;

  const __bf16* pa0 = A  + (long)(m0 + (lane & 15)) * Kd + ((lane >> 4) << 3);
  const __bf16* pa1 = pa0 + 16L * Kd;
  const __bf16* pb0 = Bw + (long)(n0 + (lane & 15)) * Kd + ((lane >> 4) << 3);
  const __bf16* pb1 = pb0 + 16L * Kd;

  v8f acc00 = {}, acc01 = {}, acc10 = {}, acc11 = {};
  for (int k0 = 0; k0 < Kd; k0 += 32) {
    v16bf a0 = gmem_frag(pa0 + k0);
    v16bf a1 = gmem_frag(pa1 + k0);
    v16bf b0 = gmem_frag(pb0 + k0);
    v16bf b1 = gmem_frag(pb1 + k0);
    acc00 = wmma_bf16(a0, b0, acc00);
    acc01 = wmma_bf16(a0, b1, acc01);
    acc10 = wmma_bf16(a1, b0, acc10);
    acc11 = wmma_bf16(a1, b1, acc11);
  }
  int col = lane & 15;
  int rb  = (lane >> 4) << 3;
  for (int r = 0; r < 8; r++) {
    C[(long)(m0 + rb + r)      * N + n0 + col]      = acc00[r];
    C[(long)(m0 + rb + r)      * N + n0 + 16 + col] = acc01[r];
    C[(long)(m0 + 16 + rb + r) * N + n0 + col]      = acc10[r];
    C[(long)(m0 + 16 + rb + r) * N + n0 + 16 + col] = acc11[r];
  }
}

// ---------------------------------------------------------------------------
// Causal depthwise conv (W=4) + bias + SiLU -> shift (f32 + bf16)
// ---------------------------------------------------------------------------
__global__ void conv_silu(const float* __restrict__ x, const float* __restrict__ cw,
                          const float* __restrict__ cb, float* __restrict__ sf,
                          __bf16* __restrict__ sb) {
  long idx = blockIdx.x * 256L + threadIdx.x;  // total = 4*2048*2048
  int  di  = (int)(idx & 2047);
  long bt  = idx >> 11;
  int  t   = (int)(bt & 2047);
  float acc = cb[di];
#pragma unroll
  for (int j = 0; j < 4; j++) {
    int tt = t - 3 + j;
    if (tt >= 0) acc += x[(bt + (long)(tt - t)) * 2048L + di] * cw[di * 4 + j];
  }
  float s = siluf(acc);
  sf[idx] = s;
  sb[idx] = to_bf16(s);
}

// ---------------------------------------------------------------------------
// v = sigmoid(t1 @ ig_w2^T + ig_b2) * x, written pre-transposed bf16 as
// vt[b][chunk][v(2048)][i(64)] so each scan block's per-chunk V^T tile is a
// single contiguous 8KB region (async-load friendly).
// ---------------------------------------------------------------------------
__global__ void v_gate(const float* __restrict__ x, const float* __restrict__ t1,
                       const float* __restrict__ ig2, const float* __restrict__ igb,
                       __bf16* __restrict__ vt) {
  long idx = blockIdx.x * 256L + threadIdx.x;
  int  di  = (int)(idx & 2047);
  long bt  = idx >> 11;
  float acc = igb[di];
#pragma unroll
  for (int r = 0; r < 16; r++) acc += t1[bt * 16 + r] * ig2[di * 16 + r];
  float v = sigmf(acc) * x[idx];

  int t = (int)(bt & 2047);
  long b = bt >> 11;
  int c = t >> 6, i = t & 63;
  vt[((b * 32 + c) * 2048L + di) * 64 + i] = to_bf16(v);
}

// ---------------------------------------------------------------------------
// Epilogue on fused q/k/g/tau projection
// ---------------------------------------------------------------------------
__global__ void qkgt_epilogue(const float* __restrict__ qkgt, const float* __restrict__ g_b,
                              const float* __restrict__ tau_b, float* __restrict__ q_s,
                              float* __restrict__ kn, float* __restrict__ gl) {
  int  rib = threadIdx.x >> 6;
  int  kk  = threadIdx.x & 63;
  long bt  = blockIdx.x * 4L + rib;
  __shared__ float red[256];

  const float* rowp = qkgt + bt * 256;
  float qv   = rowp[kk] * 0.125f;               // SCALE = 64^-0.5
  float kv   = rowp[64 + kk];
  float graw = rowp[128 + kk] + g_b[kk];
  float traw = rowp[192 + kk] + tau_b[kk];
  float gg   = (graw > 20.f) ? graw : log1pf(__expf(graw));   // softplus
  float tau  = sigmf(traw);
  float itg  = __expf(tau * __logf(fmaxf(gg, 1e-30f)));       // g^tau
  float rt   = -gg * tau;

  red[threadIdx.x] = kv * kv;
  __syncthreads();
  for (int s = 32; s > 0; s >>= 1) {
    if (kk < s) red[threadIdx.x] += red[threadIdx.x + s];
    __syncthreads();
  }
  float nrm = sqrtf(red[rib << 6]);
  float knv = kv / fmaxf(nrm, 1e-12f) * itg;

  q_s[bt * 64 + kk] = qv;
  kn [bt * 64 + kk] = knv;
  gl [bt * 64 + kk] = rt;
}

// ---------------------------------------------------------------------------
// GLA chunked scan. Grid = B(4) * Vtiles(32) blocks of 256 threads (8 waves).
// State slice [64 keys x 64 values] lives in per-wave fp32 WMMA accumulators
// across all 32 chunks. V^T tiles stream global->LDS with ASYNCcnt-tracked
// global_load_async_to_lds_b128 (CDNA5 async copy), overlapped with the
// cumulative-decay staging work, then drained with s_wait_asynccnt 0.
// ---------------------------------------------------------------------------
__global__ void gla_scan(const float* __restrict__ qs, const float* __restrict__ kn,
                         const float* __restrict__ gl, const __bf16* __restrict__ vt,
                         float* __restrict__ o) {
  const int T = 2048;
  int b    = blockIdx.x >> 5;
  int v0   = (blockIdx.x & 31) << 6;
  int tid  = threadIdx.x;
  int lane = tid & 31;
  int w    = tid >> 5;

  __shared__ float ls_btot[64];
  __shared__ __align__(16) __bf16 ls_qb [4096];  // Qb[i][k]   = q*exp(b)
  __shared__ __align__(16) __bf16 ls_ki [4096];  // Kinv[j][k] = k*exp(-b)
  __shared__ __align__(16) __bf16 ls_kdT[4096];  // Kd^T[k][i] = k*exp(btot-b)
  __shared__ __align__(16) __bf16 ls_vT [4096];  // V^T[v][i]  (async-filled)
  __shared__ __align__(16) __bf16 ls_sT [4096];  // S^T[v][k] snapshot (bf16)
  __shared__ __align__(16) __bf16 ls_A  [4096];  // masked intra-chunk A[i][j]

  // persistent fp32 state, 2 tiles per wave (16x16 each) in C-fragment layout
  v8f Sreg[2];
  for (int it = 0; it < 2; it++)
    for (int r = 0; r < 8; r++) Sreg[it][r] = 0.f;

  int tm0[2], tn0[2];
  for (int it = 0; it < 2; it++) {
    int tt2 = w * 2 + it;
    tm0[it] = (tt2 >> 2) << 4;
    tn0[it] = (tt2 & 3) << 4;
  }
  int col16 = lane & 15;
  int rb8   = (lane >> 4) << 3;

  // per-thread async-copy addressing: 16 bf16 (32B) per thread per chunk
  unsigned lds_voff = (unsigned)(unsigned long long)(&ls_vT[tid * 16]);

  for (int c = 0; c < 32; c++) {
    long t0 = (long)b * T + (long)c * 64;

    // --- kick off async V^T tile copy (contiguous 8KB region) ---
    {
      const __bf16* gsrc = vt + ((long)(b * 32 + c) * 2048 + v0) * 64 + tid * 16;
      asm volatile(
          "global_load_async_to_lds_b128 %0, %1, off\n\t"
          "global_load_async_to_lds_b128 %0, %1, off offset:16"
          :: "v"(lds_voff), "v"(gsrc) : "memory");
    }

    // --- snapshot S -> bf16 sT (transposed) while the copy is in flight ---
    for (int it = 0; it < 2; it++) {
      int m0 = tm0[it], n0 = tn0[it];
      int vc = n0 + col16;
      for (int r = 0; r < 8; r++) {
        int kr = m0 + rb8 + r;
        ls_sT[vc * 64 + kr] = to_bf16(Sreg[it][r]);
      }
    }

    // --- cumulative decay + gated q/k staging (threads 0..63) ---
    if (tid < 64) {
      int kk = tid;
      float s = 0.f;
      for (int i = 0; i < 64; i++) {
        s += gl[(t0 + i) * 64 + kk];
        float qv = qs[(t0 + i) * 64 + kk];
        float kv = kn[(t0 + i) * 64 + kk];
        ls_qb[i * 64 + kk] = to_bf16(qv * __expf(s));
        ls_ki[i * 64 + kk] = to_bf16(kv * __expf(-s));
      }
      ls_btot[kk] = s;
      float btot = s;
      s = 0.f;
      for (int i = 0; i < 64; i++) {
        s += gl[(t0 + i) * 64 + kk];
        float kv = kn[(t0 + i) * 64 + kk];
        ls_kdT[kk * 64 + i] = to_bf16(kv * __expf(btot - s));
      }
    }

    // drain async copy, then make all LDS staging visible
    asm volatile("s_wait_asynccnt 0" ::: "memory");
    __syncthreads();

    // --- A = mask(Qb @ Kinv^T)  (16 tiles, 2 per wave) ---
    for (int it = 0; it < 2; it++) {
      int m0 = tm0[it], n0 = tn0[it];
      v8f acc = {};
      for (int k0 = 0; k0 < 64; k0 += 32) {
        v16bf fa = lds_frag64(ls_qb, m0, k0, lane);
        v16bf fb = lds_frag64(ls_ki, n0, k0, lane);
        acc = wmma_bf16(fa, fb, acc);
      }
      int col = n0 + col16;
      for (int r = 0; r < 8; r++) {
        int row = m0 + rb8 + r;
        float av = (row >= col) ? acc[r] : 0.f;   // causal mask j<=i
        ls_A[row * 64 + col] = to_bf16(av);
      }
    }
    __syncthreads();

    // --- O = Qb @ S + A @ V ; state update S = diag(e^btot)S + Kd^T V ---
    for (int it = 0; it < 2; it++) {
      int m0 = tm0[it], n0 = tn0[it];
      int col = n0 + col16;

      v8f acc = {};
      for (int k0 = 0; k0 < 64; k0 += 32) {   // inter-chunk: Qb @ S
        acc = wmma_bf16(lds_frag64(ls_qb, m0, k0, lane),
                        lds_frag64(ls_sT, n0, k0, lane), acc);
      }
      for (int k0 = 0; k0 < 64; k0 += 32) {   // intra-chunk: A @ V
        acc = wmma_bf16(lds_frag64(ls_A,  m0, k0, lane),
                        lds_frag64(ls_vT, n0, k0, lane), acc);
      }
      for (int r = 0; r < 8; r++)
        o[(t0 + m0 + rb8 + r) * 2048L + v0 + col] = acc[r];

      v8f acc2 = {};
      for (int k0 = 0; k0 < 64; k0 += 32) {   // Kd^T @ V
        acc2 = wmma_bf16(lds_frag64(ls_kdT, m0, k0, lane),
                         lds_frag64(ls_vT,  n0, k0, lane), acc2);
      }
      for (int r = 0; r < 8; r++) {
        int kr = m0 + rb8 + r;
        Sreg[it][r] = __expf(ls_btot[kr]) * Sreg[it][r] + acc2[r];
      }
    }
    __syncthreads();
  }
}

// ---------------------------------------------------------------------------
// Residual + SiLU gate + RMSNorm, output bf16 for the down GEMM.
// ---------------------------------------------------------------------------
__global__ void post_norm(const float* __restrict__ o, const float* __restrict__ shift,
                          const float* __restrict__ gatef, const float* __restrict__ resw,
                          const float* __restrict__ normw, __bf16* __restrict__ xgn) {
  long bt = blockIdx.x;
  __shared__ float red[256];
  float xg[8];
  float ss = 0.f;
#pragma unroll
  for (int i = 0; i < 8; i++) {
    int di = threadIdx.x + i * 256;
    long idx = bt * 2048L + di;
    float ov = o[idx] + shift[idx] * resw[di];
    float x  = ov * siluf(gatef[idx]);
    xg[i] = x;
    ss += x * x;
  }
  red[threadIdx.x] = ss;
  __syncthreads();
  for (int s = 128; s > 0; s >>= 1) {
    if (threadIdx.x < s) red[threadIdx.x] += red[threadIdx.x + s];
    __syncthreads();
  }
  float rms = rsqrtf(red[0] / 2048.f + 1e-5f);
#pragma unroll
  for (int i = 0; i < 8; i++) {
    int di = threadIdx.x + i * 256;
    xgn[bt * 2048L + di] = to_bf16(xg[i] * rms * normw[di]);
  }
}

// ---------------------------------------------------------------------------
// Host-side pipeline
// ---------------------------------------------------------------------------
extern "C" void kernel_launch(void* const* d_in, const int* in_sizes, int n_in,
                              void* d_out, int out_size, void* d_ws, size_t ws_size,
                              hipStream_t stream) {
  (void)in_sizes; (void)n_in; (void)out_size; (void)ws_size;
  const float* hidden = (const float*)d_in[0];
  const float* up_w   = (const float*)d_in[1];
  const float* gate_w = (const float*)d_in[2];
  const float* conv_w = (const float*)d_in[3];
  const float* conv_b = (const float*)d_in[4];
  const float* q_w    = (const float*)d_in[5];
  const float* k_w    = (const float*)d_in[6];
  const float* g_w    = (const float*)d_in[7];
  const float* g_b    = (const float*)d_in[8];
  const float* tau_w  = (const float*)d_in[9];
  const float* tau_b  = (const float*)d_in[10];
  const float* ig_w1  = (const float*)d_in[11];
  const float* ig_w2  = (const float*)d_in[12];
  const float* ig_b2  = (const float*)d_in[13];
  const float* resw   = (const float*)d_in[14];
  const float* normw  = (const float*)d_in[15];
  const float* down_w = (const float*)d_in[16];
  float* out = (float*)d_out;

  char* ws = (char*)d_ws;
  size_t off = 0;
  auto take = [&](size_t bytes) -> char* {
    char* p = ws + off;
    off += (bytes + 255) & ~(size_t)255;
    return p;
  };

  __bf16* hid_bf   = (__bf16*)take(8388608L * 2);   // [8192,1024]
  __bf16* wup_bf   = (__bf16*)take(2097152L * 2);   // [2048,1024]
  __bf16* wgate_bf = (__bf16*)take(2097152L * 2);
  __bf16* wqkgt_bf = (__bf16*)take(524288L * 2);    // [256,2048] concat q/k/g/tau
  __bf16* wig1_bf  = (__bf16*)take(32768L * 2);     // [16,2048]
  __bf16* wdown_bf = (__bf16*)take(2097152L * 2);   // [1024,2048]
  float*  x_f      = (float*)take(16777216L * 4);   // [8192,2048]  (reused as o)
  float*  gate_f   = (float*)take(16777216L * 4);
  float*  shift_f  = (float*)take(16777216L * 4);
  __bf16* shift_bf = (__bf16*)take(16777216L * 2);
  __bf16* x_bf     = (__bf16*)take(16777216L * 2);  // (reused as xgn)
  float*  t1_f     = (float*)take(131072L * 4);     // [8192,16]
  __bf16* vt_bf    = (__bf16*)take(16777216L * 2);  // v, pre-transposed [b][c][v][i]
  float*  qkgt_f   = (float*)take(2097152L * 4);    // [8192,256]
  float*  qs_f     = (float*)take(524288L * 4);     // [8192,64]
  float*  kn_f     = (float*)take(524288L * 4);
  float*  gl_f     = (float*)take(524288L * 4);
  float*  o_f      = x_f;     // x dead after v_gate; scan output aliases it
  __bf16* xgn_bf   = x_bf;    // x_bf dead after t1 GEMM; post output aliases it

  // 1) precision conversions
  f32_to_bf16<<<8192, 256, 0, stream>>>(hidden, hid_bf, 8388608L);
  f32_to_bf16<<<2048, 256, 0, stream>>>(up_w, wup_bf, 2097152L);
  f32_to_bf16<<<2048, 256, 0, stream>>>(gate_w, wgate_bf, 2097152L);
  f32_to_bf16<<<128, 256, 0, stream>>>(q_w,   wqkgt_bf,           131072L);
  f32_to_bf16<<<128, 256, 0, stream>>>(k_w,   wqkgt_bf + 131072,  131072L);
  f32_to_bf16<<<128, 256, 0, stream>>>(g_w,   wqkgt_bf + 262144,  131072L);
  f32_to_bf16<<<128, 256, 0, stream>>>(tau_w, wqkgt_bf + 393216,  131072L);
  f32_to_bf16<<<32, 256, 0, stream>>>(ig_w1, wig1_bf, 32768L);
  f32_to_bf16<<<2048, 256, 0, stream>>>(down_w, wdown_bf, 2097152L);

  // 2) x = hidden @ up^T ; final_gate = hidden @ gate^T   (WMMA, 2x2 blocked)
  gemm_bf16_wmma2<<<2048, 256, 0, stream>>>(hid_bf, wup_bf,   x_f,    8192, 2048, 1024);
  gemm_bf16_wmma2<<<2048, 256, 0, stream>>>(hid_bf, wgate_bf, gate_f, 8192, 2048, 1024);

  // 3) causal conv + SiLU
  conv_silu<<<65536, 256, 0, stream>>>(x_f, conv_w, conv_b, shift_f, shift_bf);

  // 4) v = sigmoid(x@ig1^T@ig2^T + b)*x, written transposed bf16
  f32_to_bf16<<<8192, 256, 0, stream>>>(x_f, x_bf, 16777216L);
  gemm_bf16_wmma<<<64, 256, 0, stream>>>(x_bf, wig1_bf, t1_f, 8192, 16, 2048);
  v_gate<<<65536, 256, 0, stream>>>(x_f, t1_f, ig_w2, ig_b2, vt_bf);

  // 5) fused q/k/g/tau projection + gating epilogue
  gemm_bf16_wmma2<<<256, 256, 0, stream>>>(shift_bf, wqkgt_bf, qkgt_f, 8192, 256, 2048);
  qkgt_epilogue<<<2048, 256, 0, stream>>>(qkgt_f, g_b, tau_b, qs_f, kn_f, gl_f);

  // 6) GLA chunked scan (state in WMMA fp32 accumulators, async V staging)
  gla_scan<<<128, 256, 0, stream>>>(qs_f, kn_f, gl_f, vt_bf, o_f);

  // 7) residual + SiLU gate + RMSNorm
  post_norm<<<8192, 256, 0, stream>>>(o_f, shift_f, gate_f, resw, normw, xgn_bf);

  // 8) down projection -> d_out
  gemm_bf16_wmma2<<<1024, 256, 0, stream>>>(xgn_bf, wdown_bf, out, 8192, 1024, 2048);
}
